// H4AttentionLayer_52707838656618
// MI455X (gfx1250) — matrix operand order repaired and checked by reading, not compiled, not run on hardware
//
#include <hip/hip_runtime.h>
#include <hip/hip_bf16.h>

typedef __attribute__((ext_vector_type(16))) _Float16 v16h;
typedef __attribute__((ext_vector_type(8)))  _Float16 v8h;
typedef __attribute__((ext_vector_type(8)))  float    v8f;
typedef __attribute__((ext_vector_type(2)))  float    v2f;
typedef __attribute__((ext_vector_type(4)))  unsigned int u32x4;
typedef __attribute__((ext_vector_type(8)))  int      i32x8;
typedef __attribute__((ext_vector_type(4)))  int      i32x4;

#define DMODEL 1024
#define NHEADS 16
#define DHEAD  4
#define DVAL   64
#define BATCH  2
#define TLEN   2048
#define MTOK   (BATCH*TLEN)

#if __has_builtin(__builtin_amdgcn_tensor_load_to_lds) && __has_builtin(__builtin_amdgcn_s_wait_tensorcnt)
#define HAVE_TDM 1
#else
#define HAVE_TDM 0
#endif

// ---- compile-time probe (shows in stderr; no codegen effect) ----
#define CDNA5_STR2(x) #x
#define CDNA5_STR(x) CDNA5_STR2(x)
#pragma message("CDNA5 probe: clang_major=" CDNA5_STR(__clang_major__) " HAVE_TDM=" CDNA5_STR(HAVE_TDM))

// ---------------- elementwise converts ----------------
__global__ void cvt_f32_to_f16(const float* __restrict__ in, _Float16* __restrict__ out, int n) {
    int i = blockIdx.x * blockDim.x + threadIdx.x;
    if (i < n) out[i] = (_Float16)in[i];
}

// W is (N x K) row-major fp32; out is W^T (K x N) row-major f16
__global__ void tcvt_f32_to_f16(const float* __restrict__ in, _Float16* __restrict__ out, int N, int K) {
    int i = blockIdx.x * blockDim.x + threadIdx.x;
    if (i < N * K) {
        int k = i / N, n = i % N;
        out[i] = (_Float16)in[(size_t)n * K + k];
    }
}

// ---------------- TDM-staged f16 WMMA GEMM (N multiple of 128, M multiple of 32) ----------------
// Defined FIRST among WMMA kernels so the disasm snippet window lands on its inner loop.
#define GBN 128
#define GPITCH 136            /* halves per LDS row: 128 + 8 pad (TDM pads 4 dwords / 64 dwords) */
#define GBUFH (32 * GPITCH)   /* halves per LDS buffer */

#if HAVE_TDM
__device__ __forceinline__ void tdm_load_b(unsigned lds_byte_addr, const _Float16* gsrc, int N) {
    unsigned long long ga = (unsigned long long)(size_t)gsrc;
    u32x4 g0;
    g0[0] = 1u;                                                   // count=1, user descriptor
    g0[1] = lds_byte_addr;                                        // LDS dest (bytes)
    g0[2] = (unsigned)ga;                                         // global addr lo
    g0[3] = (unsigned)((ga >> 32) & 0x01ffffffu) | (2u << 30);    // global addr hi | type=2
    i32x8 g1;
    g1[0] = (int)((1u << 16) | (1u << 20) | (5u << 22) | (3u << 25)); // data_size=2B, pad_en, 64 dw interval, 4 dw pad
    g1[1] = 0;                                                    // tensor_dim0 lo16 (of 2^30) = 0
    g1[2] = 0x4000;                                               // tensor_dim0 hi; dim1 lo16 = 0
    g1[3] = (int)(0x4000u | ((unsigned)GBN << 16));               // tensor_dim1 hi; tile_dim0 = 128
    g1[4] = 32;                                                   // tile_dim1 = 32 rows
    g1[5] = N;                                                    // tensor_dim0_stride (elements)
    g1[6] = 0;
    g1[7] = 0;
    i32x4 z = {};
#if defined(__clang_major__) && (__clang_major__ >= 23)
    i32x8 z8 = {};
    __builtin_amdgcn_tensor_load_to_lds(g0, g1, z, z, z8, 0);
#else
    __builtin_amdgcn_tensor_load_to_lds(g0, g1, z, z, 0);
#endif
}
#endif

__global__ void gemm_f16_wmma_tdm(const _Float16* __restrict__ A,
                                  const _Float16* __restrict__ Bt,
                                  float* __restrict__ C,
                                  int M, int N, int K) {
    __shared__ _Float16 smem[2 * GBUFH];
    int tid = threadIdx.x;
    int lane = tid & 31, wave = tid >> 5;
    int nBlkN = N / GBN;
    int bm = blockIdx.x / nBlkN, bn = blockIdx.x % nBlkN;
    int mrow = wave >> 2, ncol = wave & 3;
    int m0 = bm * 32 + mrow * 16;
    int n0 = bn * GBN + ncol * 32;
    int hl = lane >> 4, lr = lane & 15;
    const _Float16* arow = A + (size_t)(m0 + lr) * K;
    int steps = K >> 5;
#if HAVE_TDM
    unsigned ldsbase = (unsigned)(size_t)(void*)smem;   // LDS aperture: addr[31:0] is the LDS offset
    if (wave == 0) tdm_load_b(ldsbase, Bt + (size_t)bn * GBN, N);
#endif
    v8f acc0 = {}, acc1 = {};
    for (int i = 0; i < steps; ++i) {
        int kk = i << 5;
#if HAVE_TDM
        if (wave == 0) {
            if (i + 1 < steps) {
                tdm_load_b(ldsbase + (unsigned)(((i + 1) & 1) * GBUFH * 2),
                           Bt + (size_t)(kk + 32) * N + (size_t)bn * GBN, N);
                __builtin_amdgcn_s_wait_tensorcnt(1);   // previous slab complete
            } else {
                __builtin_amdgcn_s_wait_tensorcnt(0);
            }
        }
        __syncthreads();
#else
        {   // cooperative staging fallback
            const _Float16* src = Bt + (size_t)kk * N + (size_t)bn * GBN;
            for (int e = tid; e < 32 * (GBN / 8); e += 256) {
                int row = e / (GBN / 8), c8 = (e % (GBN / 8)) * 8;
                *(v8h*)(&smem[(i & 1) * GBUFH + row * GPITCH + c8]) = *(const v8h*)(src + (size_t)row * N + c8);
            }
            __syncthreads();
        }
#endif
        const _Float16* bbuf = smem + (i & 1) * GBUFH;
        v8h a0 = *(const v8h*)(arow + kk + 8 * hl);
        v8h a1 = *(const v8h*)(arow + kk + 16 + 8 * hl);
        const _Float16* brow = bbuf + (8 * hl) * GPITCH + ncol * 32 + lr;  // lane%16 = N column, elements walk K rows
        v8h b00, b01, b10, b11;
#pragma unroll
        for (int e = 0; e < 8; ++e) {
            b00[e] = brow[e * GPITCH];       b01[e] = brow[(16 + e) * GPITCH];
            b10[e] = brow[e * GPITCH + 16];  b11[e] = brow[(16 + e) * GPITCH + 16];
        }
        v16h av, bv0, bv1;
#pragma unroll
        for (int e = 0; e < 8; ++e) {
            av[e] = a0[e]; av[8 + e] = a1[e];
            bv0[e] = b00[e]; bv0[8 + e] = b01[e];
            bv1[e] = b10[e]; bv1[8 + e] = b11[e];
        }
        acc0 = __builtin_amdgcn_wmma_f32_16x16x32_f16(false, av, false, bv0, (short)0, acc0, false, false);
        acc1 = __builtin_amdgcn_wmma_f32_16x16x32_f16(false, av, false, bv1, (short)0, acc1, false, false);
        __syncthreads();   // all waves done with this buffer before it is re-filled
    }
#pragma unroll
    for (int r = 0; r < 8; ++r) {
        size_t row = (size_t)(m0 + r + 8 * hl) * N;
        C[row + n0 + lr]      = acc0[r];
        C[row + n0 + 16 + lr] = acc1[r];
    }
}

// V (B*T, H*64) f32 -> Vt ((b*H+h)*64+dv, T) f32  (key-contiguous for attention B-operand)
__global__ void transpose_v(const float* __restrict__ V, float* __restrict__ Vt) {
    int tid = blockIdx.x * blockDim.x + threadIdx.x;
    if (tid >= MTOK * DMODEL) return;
    int t = tid % TLEN;
    int rest = tid / TLEN;          // b*1024 + h*64 + dv
    int col = rest % DMODEL;
    int b = rest / DMODEL;
    Vt[tid] = V[((size_t)(b * TLEN) + t) * DMODEL + col];
}

// ---------------- simple f16 WMMA GEMM (narrow Q/K projections, N=64) ----------------
__global__ void gemm_f16_wmma(const _Float16* __restrict__ A,
                              const _Float16* __restrict__ Bt,
                              float* __restrict__ C,
                              int M, int N, int K) {
    int lane = threadIdx.x & 31;
    int wave = threadIdx.x >> 5;
    int tilesN = N >> 4;
    int totalTiles = (M >> 4) * tilesN;
    int tile = blockIdx.x * 8 + wave;
    if (tile >= totalTiles) return;
    int m0 = (tile / tilesN) << 4;
    int n0 = (tile % tilesN) << 4;
    int hl = lane >> 4, lr = lane & 15;

    const _Float16* arow = A + (size_t)(m0 + lr) * K;
    v8f acc = {};
    for (int kk = 0; kk < K; kk += 32) {
        v8h a0 = *(const v8h*)(arow + kk + 8 * hl);
        v8h a1 = *(const v8h*)(arow + kk + 16 + 8 * hl);
        const _Float16* brow = Bt + (size_t)(kk + 8 * hl) * N + n0 + lr;
        v8h b0, b1;
#pragma unroll
        for (int i = 0; i < 8; ++i) { b0[i] = brow[(size_t)i * N]; b1[i] = brow[(size_t)(16 + i) * N]; }
        v16h av, bv;
#pragma unroll
        for (int i = 0; i < 8; ++i) { av[i] = a0[i]; av[8 + i] = a1[i]; bv[i] = b0[i]; bv[8 + i] = b1[i]; }
        acc = __builtin_amdgcn_wmma_f32_16x16x32_f16(false, av, false, bv, (short)0, acc, false, false);
    }
#pragma unroll
    for (int r = 0; r < 8; ++r)
        C[(size_t)(m0 + r + 8 * hl) * N + n0 + lr] = acc[r];
}

// ---------------- Q/K post-processing: normalize, nudge, chamber bonus ----------------
__global__ void qk_post(const float* __restrict__ Qraw, const float* __restrict__ Kraw,
                        const float* __restrict__ Wn, const float* __restrict__ SR,
                        const float* __restrict__ CB,
                        float* __restrict__ Qn, float* __restrict__ Kn, float* __restrict__ KB) {
    int tid = blockIdx.x * blockDim.x + threadIdx.x;
    if (tid >= MTOK * NHEADS) return;
    int m = tid >> 4, h = tid & 15;
    const float* q = Qraw + (size_t)m * 64 + h * 4;
    const float* k = Kraw + (size_t)m * 64 + h * 4;
    float qv[4], kv[4], qn2 = 1e-24f, kn2 = 1e-24f;
#pragma unroll
    for (int d = 0; d < 4; ++d) { qv[d] = q[d]; kv[d] = k[d]; qn2 += qv[d] * qv[d]; kn2 += kv[d] * kv[d]; }
    float qi = rsqrtf(qn2), ki = rsqrtf(kn2);
    float qe[4], qe2 = 1e-24f;
#pragma unroll
    for (int e = 0; e < 4; ++e) {
        float s = 0.f;
#pragma unroll
        for (int d = 0; d < 4; ++d) s += qv[d] * qi * Wn[h * 16 + d * 4 + e];
        qe[e] = s; qe2 += s * s;
    }
    float qei = rsqrtf(qe2) * 0.5f;   // fold SCALE = 1/sqrt(4)
#pragma unroll
    for (int e = 0; e < 4; ++e) Qn[(size_t)m * 64 + h * 4 + e] = qe[e] * qei;
#pragma unroll
    for (int d = 0; d < 4; ++d) Kn[(size_t)m * 64 + h * 4 + d] = kv[d] * ki;
    float sg[4];
#pragma unroll
    for (int r = 0; r < 4; ++r) {
        float dt = 0.f;
#pragma unroll
        for (int d = 0; d < 4; ++d) dt += kv[d] * ki * SR[r * 4 + d];
        sg[r] = 1.0f / (1.0f + __expf(-3.0f * dt));
    }
    float bonus = 0.f;
#pragma unroll
    for (int c = 0; c < 16; ++c) {
        float w = 1.f;
#pragma unroll
        for (int j = 0; j < 4; ++j) { float s = sg[j]; w *= ((c >> j) & 1) ? s : (1.f - s); }
        bonus += w * CB[h * 16 + c];
    }
    KB[(size_t)m * 16 + h] = bonus;
}

// ---------------- flash attention with f32 16x16x4 WMMA ----------------
__global__ void attn_h4(const float* __restrict__ Qn, const float* __restrict__ Kn,
                        const float* __restrict__ KB, const float* __restrict__ Vt,
                        _Float16* __restrict__ Oh) {
    const int QT = TLEN / 16;
    int lane = threadIdx.x & 31;
    int wave = threadIdx.x >> 5;
    int task = blockIdx.x * 8 + wave;
    if (task >= BATCH * NHEADS * QT) return;
    int qt = task % QT;
    int bh = task / QT;
    int h = bh % NHEADS, b = bh / NHEADS;
    int hl = lane >> 4, lr = lane & 15;
    int q0 = qt * 16;

    size_t qbase = ((size_t)(b * TLEN) + q0 + lr) * 64 + h * 4 + 2 * hl;
    v2f qb; qb[0] = Qn[qbase]; qb[1] = Qn[qbase + 1];

    float mprev = -INFINITY, lsum = 0.f;
    v8f O0 = {}, O1 = {}, O2 = {}, O3 = {};
    const float* vt = Vt + ((size_t)(b * NHEADS + h) * DVAL) * TLEN;  // vt[col*T + key]

    for (int kb0 = 0; kb0 <= q0; kb0 += 16) {
        size_t kbase = ((size_t)(b * TLEN) + kb0 + lr) * 64 + h * 4 + 2 * hl;
        v2f ka; ka[0] = Kn[kbase]; ka[1] = Kn[kbase + 1];
        v8f sc;
#pragma unroll
        for (int r = 0; r < 8; ++r)
            sc[r] = KB[((size_t)(b * TLEN) + kb0 + r + 8 * hl) * NHEADS + h];
        sc = __builtin_amdgcn_wmma_f32_16x16x4_f32(false, ka, false, qb, (short)0, sc, false, false);
        if (kb0 == q0) {
#pragma unroll
            for (int r = 0; r < 8; ++r) if ((r + 8 * hl) > lr) sc[r] = -INFINITY;
        }
        float mloc = sc[0];
#pragma unroll
        for (int r = 1; r < 8; ++r) mloc = fmaxf(mloc, sc[r]);
        mloc = fmaxf(mloc, __shfl_xor(mloc, 16, 32));
        float mnew = fmaxf(mprev, mloc);
        float alpha = __expf(mprev - mnew);
        float p[8], rs = 0.f;
#pragma unroll
        for (int r = 0; r < 8; ++r) { p[r] = __expf(sc[r] - mnew); rs += p[r]; }
        rs += __shfl_xor(rs, 16, 32);
        lsum = lsum * alpha + rs;
        mprev = mnew;
        float ar[8];
#pragma unroll
        for (int r = 0; r < 8; ++r) ar[r] = __shfl(alpha, r + 8 * hl, 32);
#pragma unroll
        for (int r = 0; r < 8; ++r) { O0[r] *= ar[r]; O1[r] *= ar[r]; O2[r] *= ar[r]; O3[r] *= ar[r]; }
        float po[8];
#pragma unroll
        for (int r = 0; r < 8; ++r) po[r] = __shfl_xor(p[r], 16, 32);
        v2f A0, A1, A2, A3;
        if (hl == 0) {
            A0[0] = p[0];  A0[1] = p[1];  A1[0] = p[4];  A1[1] = p[5];
            A2[0] = po[0]; A2[1] = po[1]; A3[0] = po[4]; A3[1] = po[5];
        } else {
            A0[0] = po[2]; A0[1] = po[3]; A1[0] = po[6]; A1[1] = po[7];
            A2[0] = p[2];  A2[1] = p[3];  A3[0] = p[6];  A3[1] = p[7];
        }
#pragma unroll
        for (int j = 0; j < 4; ++j) {
            const float* vc = vt + (size_t)(j * 16 + lr) * TLEN;   // value column, keys contiguous
            v8f* Oj = (j == 0) ? &O0 : (j == 1) ? &O1 : (j == 2) ? &O2 : &O3;
            v2f Bv;
            Bv = *(const v2f*)(vc + kb0 + 0 + 2 * hl);
            *Oj = __builtin_amdgcn_wmma_f32_16x16x4_f32(false, A0, false, Bv, (short)0, *Oj, false, false);
            Bv = *(const v2f*)(vc + kb0 + 4 + 2 * hl);
            *Oj = __builtin_amdgcn_wmma_f32_16x16x4_f32(false, A1, false, Bv, (short)0, *Oj, false, false);
            Bv = *(const v2f*)(vc + kb0 + 8 + 2 * hl);
            *Oj = __builtin_amdgcn_wmma_f32_16x16x4_f32(false, A2, false, Bv, (short)0, *Oj, false, false);
            Bv = *(const v2f*)(vc + kb0 + 12 + 2 * hl);
            *Oj = __builtin_amdgcn_wmma_f32_16x16x4_f32(false, A3, false, Bv, (short)0, *Oj, false, false);
        }
    }
    float inv[8];
#pragma unroll
    for (int r = 0; r < 8; ++r) inv[r] = 1.0f / __shfl(lsum, r + 8 * hl, 32);
    size_t obase = ((size_t)(b * TLEN) + q0) * (NHEADS * DVAL) + h * DVAL;
#pragma unroll
    for (int r = 0; r < 8; ++r) {
        size_t row = obase + (size_t)(r + 8 * hl) * 1024;
        Oh[row + 0 + lr]  = (_Float16)(O0[r] * inv[r]);
        Oh[row + 16 + lr] = (_Float16)(O1[r] * inv[r]);
        Oh[row + 32 + lr] = (_Float16)(O2[r] * inv[r]);
        Oh[row + 48 + lr] = (_Float16)(O3[r] * inv[r]);
    }
}

// ---------------- host-side launch ----------------
extern "C" void kernel_launch(void* const* d_in, const int* in_sizes, int n_in,
                              void* d_out, int out_size, void* d_ws, size_t ws_size,
                              hipStream_t stream) {
    const float* x    = (const float*)d_in[0];
    const float* Wq   = (const float*)d_in[1];
    const float* Wk   = (const float*)d_in[2];
    const float* Wv   = (const float*)d_in[3];
    const float* Wn   = (const float*)d_in[4];
    const float* CB   = (const float*)d_in[5];
    const float* Wout = (const float*)d_in[6];
    const float* SR   = (const float*)d_in[7];
    float* out = (float*)d_out;

    char* ws = (char*)d_ws;
    size_t o = 0;
    auto alloc = [&](size_t bytes) { size_t r = o; o += (bytes + 255) & ~(size_t)255; return r; };
    size_t o_xh   = alloc((size_t)MTOK * DMODEL * 2);
    size_t o_wqT  = alloc((size_t)DMODEL * 64 * 2);
    size_t o_wkT  = alloc((size_t)DMODEL * 64 * 2);
    size_t o_wvT  = alloc((size_t)DMODEL * DMODEL * 2);
    size_t o_woT  = alloc((size_t)DMODEL * DMODEL * 2);
    size_t o_qraw = alloc((size_t)MTOK * 64 * 4);
    size_t o_kraw = alloc((size_t)MTOK * 64 * 4);
    size_t o_qn   = alloc((size_t)MTOK * 64 * 4);
    size_t o_kn   = alloc((size_t)MTOK * 64 * 4);
    size_t o_kb   = alloc((size_t)MTOK * NHEADS * 4);
    size_t o_v    = alloc((size_t)MTOK * DMODEL * 4);
    size_t o_vt   = alloc((size_t)MTOK * DMODEL * 4);
    size_t o_aoh  = alloc((size_t)MTOK * DMODEL * 2);
    (void)ws_size; (void)n_in; (void)in_sizes; (void)out_size;

    _Float16* xh  = (_Float16*)(ws + o_xh);
    _Float16* wqT = (_Float16*)(ws + o_wqT);
    _Float16* wkT = (_Float16*)(ws + o_wkT);
    _Float16* wvT = (_Float16*)(ws + o_wvT);
    _Float16* woT = (_Float16*)(ws + o_woT);
    float* qraw = (float*)(ws + o_qraw);
    float* kraw = (float*)(ws + o_kraw);
    float* qn   = (float*)(ws + o_qn);
    float* kn   = (float*)(ws + o_kn);
    float* kb   = (float*)(ws + o_kb);
    float* v    = (float*)(ws + o_v);
    float* vtb  = (float*)(ws + o_vt);
    _Float16* aoh = (_Float16*)(ws + o_aoh);

    const int TB = 256;
    cvt_f32_to_f16<<<(MTOK * DMODEL + TB - 1) / TB, TB, 0, stream>>>(x, xh, MTOK * DMODEL);
    tcvt_f32_to_f16<<<(64 * DMODEL + TB - 1) / TB, TB, 0, stream>>>(Wq, wqT, 64, DMODEL);
    tcvt_f32_to_f16<<<(64 * DMODEL + TB - 1) / TB, TB, 0, stream>>>(Wk, wkT, 64, DMODEL);
    tcvt_f32_to_f16<<<(DMODEL * DMODEL + TB - 1) / TB, TB, 0, stream>>>(Wv, wvT, DMODEL, DMODEL);
    tcvt_f32_to_f16<<<(DMODEL * DMODEL + TB - 1) / TB, TB, 0, stream>>>(Wout, woT, DMODEL, DMODEL);
    {   // narrow Q/K projections
        int tiles = (MTOK / 16) * (64 / 16);
        gemm_f16_wmma<<<(tiles + 7) / 8, TB, 0, stream>>>(xh, wqT, qraw, MTOK, 64, DMODEL);
        gemm_f16_wmma<<<(tiles + 7) / 8, TB, 0, stream>>>(xh, wkT, kraw, MTOK, 64, DMODEL);
    }
    {   // V projection: TDM-staged GEMM
        int blocks = (MTOK / 32) * (DMODEL / GBN);
        gemm_f16_wmma_tdm<<<blocks, TB, 0, stream>>>(xh, wvT, v, MTOK, DMODEL, DMODEL);
    }
    transpose_v<<<(MTOK * DMODEL + TB - 1) / TB, TB, 0, stream>>>(v, vtb);
    qk_post<<<(MTOK * NHEADS + TB - 1) / TB, TB, 0, stream>>>(qraw, kraw, Wn, SR, CB, qn, kn, kb);
    {   // flash attention
        int tasks = BATCH * NHEADS * (TLEN / 16);
        attn_h4<<<(tasks + 7) / 8, TB, 0, stream>>>(qn, kn, kb, vtb, aoh);
    }
    {   // output projection: TDM-staged GEMM -> fp32 d_out
        int blocks = (MTOK / 32) * (DMODEL / GBN);
        gemm_f16_wmma_tdm<<<blocks, TB, 0, stream>>>(aoh, woT, out, MTOK, DMODEL, DMODEL);
    }
}
